// RC_Explainer_Batch_star_71725953843767
// MI455X (gfx1250) — matrix operand order repaired and checked by reading, not compiled, not run on hardware
//
#include <hip/hip_runtime.h>
#include <hip/hip_bf16.h>

typedef __attribute__((ext_vector_type(16))) _Float16 v16h;
typedef __attribute__((ext_vector_type(8)))  float    v8f;

#define NB    64
#define DIM   128
#define HID   64
#define NLBL  4
#define RDIM  32

// Branchless ELU: always evaluate exp on clamped arg (v_exp_f32), select with v_cndmask.
__device__ __forceinline__ float elu_f(float x){
  float e = __expf(fminf(x, 0.f)) - 1.f;
  return x > 0.f ? x : e;
}

__device__ __forceinline__ unsigned enc_f32(float f){
  unsigned u = __float_as_uint(f);
  return (u & 0x80000000u) ? ~u : (u | 0x80000000u);
}
__device__ __forceinline__ float dec_f32(unsigned u){
  unsigned b = (u & 0x80000000u) ? (u & 0x7FFFFFFFu) : ~u;
  return __uint_as_float(b);
}

// ---------------- WMMA fragment helpers (wave32, v_wmma_f32_16x16x32_f16) --------
// A tile (16x32 f16): lanes 0-15 row M=lane, halves {k0+0..7, k0+16..23};
// lanes 16-31 row M=lane-16, halves {k0+8..15, k0+24..31}.
__device__ __forceinline__ v16h load_a_f16(const _Float16* buf, int stride, int k0, int lane){
  const _Float16* p = buf + (lane & 15) * stride + k0 + ((lane < 16) ? 0 : 8);
  v16h a;
#pragma unroll
  for (int i = 0; i < 8; ++i){ a[i] = p[i]; a[i + 8] = p[i + 16]; }
  return a;
}
__device__ __forceinline__ v16h load_a_f32g(const float* buf, int stride, int m0, int k0, int lane){
  const float* p = buf + (size_t)(m0 + (lane & 15)) * stride + k0 + ((lane < 16) ? 0 : 8);
  v16h a;
#pragma unroll
  for (int i = 0; i < 8; ++i){ a[i] = (_Float16)p[i]; a[i + 8] = (_Float16)p[i + 16]; }
  return a;
}
// B tile (32x16 f16) from transposed weights Wt[N][K]: lanes 0-15 col n0+lane K {k0+0..7, k0+16..23},
// lanes 16-31 col n0+lane-16 K {k0+8..15, k0+24..31} (matches A layout).
__device__ __forceinline__ v16h load_b_f16(const _Float16* Wt, int K, int n0, int k0, int lane){
  const _Float16* p = Wt + (size_t)(n0 + (lane & 15)) * K + k0 + ((lane < 16) ? 0 : 8);
  v16h b;
#pragma unroll
  for (int i = 0; i < 8; ++i){ b[i] = p[i]; b[i + 8] = p[i + 16]; }
  return b;
}
__device__ __forceinline__ v8f bias_acc(const float* bias, int n0, int lane){
  float bv = bias ? bias[n0 + (lane & 15)] : 0.f;
  v8f c;
#pragma unroll
  for (int j = 0; j < 8; ++j) c[j] = bv;
  return c;
}
// D (16x16 f32, column-per-lane) -> f16 row-major LDS
__device__ __forceinline__ void store_d_f16(_Float16* buf, int stride, int n0, v8f acc, int lane){
  int col = n0 + (lane & 15);
  int r0  = (lane < 16) ? 0 : 8;
#pragma unroll
  for (int j = 0; j < 8; ++j) buf[(r0 + j) * stride + col] = (_Float16)acc[j];
}
__device__ __forceinline__ v8f elu_v(v8f a){
#pragma unroll
  for (int j = 0; j < 8; ++j) a[j] = elu_f(a[j]);
  return a;
}
#define WMMA_F16(A,B,C) __builtin_amdgcn_wmma_f32_16x16x32_f16(false,(A),false,(B),(short)0,(C),false,false)

// ---------------- prep: Wt[n*K+k] = (f16) W[k*N+n] ----------------
__global__ void k_transpose_w(const float* __restrict__ W, _Float16* __restrict__ Wt, int K, int Nc){
  int idx = blockIdx.x * blockDim.x + threadIdx.x;
  if (idx >= K * Nc) return;
  int k = idx / Nc, n = idx - k * Nc;
  Wt[(size_t)n * K + k] = (_Float16)W[idx];
}

// ---------------- h = elu(x @ Wm1)  [N,128]@[128,64] ----------------
__global__ __launch_bounds__(128) void k_h(const float* __restrict__ x,
                                           const _Float16* __restrict__ Wm1t,
                                           float* __restrict__ h, int nnodes){
  int lane = threadIdx.x & 31;
  int tile = blockIdx.x * 4 + (threadIdx.x >> 5);
  int m0 = tile * 16;
  if (m0 >= nnodes) return;
  v16h a[4];
#pragma unroll
  for (int kt = 0; kt < 4; ++kt) a[kt] = load_a_f32g(x, DIM, m0, kt * 32, lane);
#pragma unroll
  for (int nt = 0; nt < 4; ++nt){
    v8f acc = bias_acc(nullptr, 0, lane);
#pragma unroll
    for (int kt = 0; kt < 4; ++kt)
      acc = WMMA_F16(a[kt], load_b_f16(Wm1t, DIM, nt * 16, kt * 32, lane), acc);
    acc = elu_v(acc);
    int col = nt * 16 + (lane & 15);
    int r0  = (lane < 16) ? 0 : 8;
#pragma unroll
    for (int j = 0; j < 8; ++j) h[(size_t)(m0 + r0 + j) * HID + col] = acc[j];
  }
}

// ---------------- edge scatter: agg_sf / agg_ava += h[src]; nmask ----------------
__global__ void k_scatter(const int* __restrict__ src, const int* __restrict__ dst,
                          const unsigned char* __restrict__ state, const float* __restrict__ h,
                          float* __restrict__ agg_sf, float* __restrict__ agg_ava,
                          float* __restrict__ nmask, int E){
  long tid = (long)blockIdx.x * blockDim.x + threadIdx.x;
  if (tid >= (long)E * 16) return;
  int e = (int)(tid >> 4);
  int c = ((int)tid & 15) * 4;
  int s = src[e], d = dst[e];
  bool occ = state[e] != 0;
  const float* hp = h + (size_t)s * HID + c;
  float* tgt = (occ ? agg_sf : agg_ava) + (size_t)d * HID + c;
  atomicAdd(tgt + 0, hp[0]); atomicAdd(tgt + 1, hp[1]);
  atomicAdd(tgt + 2, hp[2]); atomicAdd(tgt + 3, hp[3]);
  if (c == 0 && occ){ nmask[s] = 1.f; nmask[d] = 1.f; }
}

// ---------------- node reps + pooled sums (32 threads per node) ----------------
__global__ void k_node(const float* __restrict__ h, const float* __restrict__ agg_sf,
                       const float* __restrict__ agg_ava, const float* __restrict__ Wm2,
                       const float* __restrict__ nmask, const int* __restrict__ batch,
                       float* __restrict__ nr_ava,
                       float* __restrict__ sum_ones, float* __restrict__ sum_sf,
                       float* __restrict__ cnt_ones, float* __restrict__ cnt_sf, int nnodes){
  __shared__ float sW[HID * RDIM];
  for (int i = threadIdx.x; i < HID * RDIM; i += blockDim.x) sW[i] = Wm2[i];
  __syncthreads();
  int c = threadIdx.x & 31;
  int n = blockIdx.x * (blockDim.x >> 5) + (threadIdx.x >> 5);
  if (n >= nnodes) return;
  const float* hr = h + (size_t)n * HID;
  const float* sr = agg_sf + (size_t)n * HID;
  const float* ar = agg_ava + (size_t)n * HID;
  float p = 0.f, qs = 0.f, qa = 0.f;
#pragma unroll 4
  for (int k = 0; k < HID; ++k){
    float w = sW[k * RDIM + c];
    p += hr[k] * w; qs += sr[k] * w; qa += ar[k] * w;
  }
  float vo = elu_f(p + qs + qa);   // ew = ones
  float vs = elu_f(p + qs);        // ew = occupied
  float va = elu_f(p + qa);        // ew = available
  nr_ava[(size_t)n * RDIM + c] = va;
  int b = batch[n];
  float m = nmask[n];
  atomicAdd(&sum_ones[b * RDIM + c], vo);
  atomicAdd(&sum_sf[b * RDIM + c], vs * m);
  if (c == 0){ atomicAdd(&cnt_ones[b], 1.f); atomicAdd(&cnt_sf[b], m); }
}

// ---------------- diff = (pooled_ones - pooled_sf) @ Wg  [B,64] ----------------
__global__ void k_diff(const float* __restrict__ sum_ones, const float* __restrict__ sum_sf,
                       const float* __restrict__ cnt_ones, const float* __restrict__ cnt_sf,
                       const float* __restrict__ Wg, float* __restrict__ diff){
  int b = blockIdx.x, c = threadIdx.x;
  float co = fmaxf(cnt_ones[b], 1.f), cs = fmaxf(cnt_sf[b], 1.f);
  float s = 0.f;
#pragma unroll
  for (int k = 0; k < RDIM; ++k)
    s += (sum_ones[b * RDIM + k] / co - sum_sf[b * RDIM + k] / cs) * Wg[k * HID + c];
  diff[b * HID + c] = s;
}

// ---------------- order-preserving compaction of available edges ----------------
#define SCAN_TPB 256
#define SCAN_EPT 16
__global__ void k_count(const unsigned char* __restrict__ state, int* __restrict__ blockcnt, int E){
  __shared__ int sc[SCAN_TPB];
  int base = (blockIdx.x * SCAN_TPB + threadIdx.x) * SCAN_EPT;
  int cnt = 0;
  for (int j = 0; j < SCAN_EPT; ++j){ int e = base + j; if (e < E && state[e] == 0) cnt++; }
  sc[threadIdx.x] = cnt; __syncthreads();
  if (threadIdx.x == 0){ int t = 0; for (int i = 0; i < SCAN_TPB; ++i) t += sc[i]; blockcnt[blockIdx.x] = t; }
}
__global__ void k_scanblocks(const int* __restrict__ blockcnt, int* __restrict__ blockoff, int nblk){
  if (threadIdx.x == 0){ int acc = 0; for (int i = 0; i < nblk; ++i){ blockoff[i] = acc; acc += blockcnt[i]; } }
}
__global__ void k_compact(const unsigned char* __restrict__ state, const int* __restrict__ blockoff,
                          int* __restrict__ aidx, int E){
  __shared__ int sc[SCAN_TPB];
  int base = (blockIdx.x * SCAN_TPB + threadIdx.x) * SCAN_EPT;
  int cnt = 0;
  for (int j = 0; j < SCAN_EPT; ++j){ int e = base + j; if (e < E && state[e] == 0) cnt++; }
  sc[threadIdx.x] = cnt; __syncthreads();
  if (threadIdx.x == 0){
    int acc = blockoff[blockIdx.x];
    for (int i = 0; i < SCAN_TPB; ++i){ int t = sc[i]; sc[i] = acc; acc += t; }
  }
  __syncthreads();
  int off = sc[threadIdx.x];
  for (int j = 0; j < SCAN_EPT; ++j){ int e = base + j; if (e < E && state[e] == 0) aidx[off++] = e; }
}

// ---------------- fused edge MLP + per-label explainer (WMMA) ----------------
#define K6_WAVES 4
__global__ __launch_bounds__(128) void k_edge(
    const int* __restrict__ aidx, const int* __restrict__ src, const int* __restrict__ dst,
    const int* __restrict__ batch, const int* __restrict__ y,
    const float* __restrict__ nr_ava, const float* __restrict__ diff,
    const _Float16* __restrict__ W1t, const float* __restrict__ b1,
    const _Float16* __restrict__ W2t, const float* __restrict__ b2,
    const _Float16* __restrict__ W3t, const float* __restrict__ b3,
    const _Float16* __restrict__ We1t, const float* __restrict__ be1,
    const _Float16* __restrict__ We2t, const float* __restrict__ be2,
    const float* __restrict__ We3, const float* __restrict__ be3,
    float* __restrict__ probs, int Ea){
  __shared__ _Float16 sbuf[K6_WAVES][16][256];   // activation staging (layer outputs, h1, h2)
  __shared__ _Float16 sfeat[K6_WAVES][16][128];  // feats = [reps | diff]
  int lane = threadIdx.x & 31;
  int wv   = threadIdx.x >> 5;
  int ntiles = (Ea + 15) >> 4;
  int tile = blockIdx.x * K6_WAVES + wv;
  if (tile >= ntiles) return;
  int t0 = tile * 16;
  int r = lane & 15;
  int hsel = lane >> 4;               // 0: src half, 1: dst half
  int t = t0 + r; if (t >= Ea) t = Ea - 1;
  int e = aidx[t];
  int sn = src[e], dn = dst[e];
  int node = hsel ? dn : sn;
  int bidx = batch[sn];
  int label = y[bidx];
  _Float16* SB = &sbuf[wv][0][0];
  _Float16* SF = &sfeat[wv][0][0];
  {
    const float* npR = nr_ava + (size_t)node * RDIM;
#pragma unroll
    for (int i = 0; i < RDIM; ++i) SB[r * 256 + hsel * RDIM + i] = (_Float16)npR[i];
    const float* dR = diff + (size_t)bidx * HID + hsel * 32;
#pragma unroll
    for (int i = 0; i < 32; ++i) SF[r * 128 + 64 + hsel * 32 + i] = (_Float16)dR[i];
  }
  // layer 1: [16,64] @ W1[64,256] + b1, elu
  v16h a0 = load_a_f16(SB, 256, 0, lane);
  v16h a1 = load_a_f16(SB, 256, 32, lane);
#pragma unroll
  for (int nt = 0; nt < 16; ++nt){
    v8f acc = bias_acc(b1, nt * 16, lane);
    acc = WMMA_F16(a0, load_b_f16(W1t, 64, nt * 16, 0, lane), acc);
    acc = WMMA_F16(a1, load_b_f16(W1t, 64, nt * 16, 32, lane), acc);
    store_d_f16(SB, 256, nt * 16, elu_v(acc), lane);
  }
  // layer 2: [16,256] @ W2[256,128] + b2, elu
  v16h a2[8];
#pragma unroll
  for (int kt = 0; kt < 8; ++kt) a2[kt] = load_a_f16(SB, 256, kt * 32, lane);
#pragma unroll
  for (int nt = 0; nt < 8; ++nt){
    v8f acc = bias_acc(b2, nt * 16, lane);
#pragma unroll
    for (int kt = 0; kt < 8; ++kt)
      acc = WMMA_F16(a2[kt], load_b_f16(W2t, 256, nt * 16, kt * 32, lane), acc);
    store_d_f16(SB, 256, nt * 16, elu_v(acc), lane);
  }
  // layer 3 (linear): [16,128] @ W3[128,64] + b3 -> feats cols 0..63
  v16h a3[4];
#pragma unroll
  for (int kt = 0; kt < 4; ++kt) a3[kt] = load_a_f16(SB, 256, kt * 32, lane);
#pragma unroll
  for (int nt = 0; nt < 4; ++nt){
    v8f acc = bias_acc(b3, nt * 16, lane);
#pragma unroll
    for (int kt = 0; kt < 4; ++kt)
      acc = WMMA_F16(a3[kt], load_b_f16(W3t, 128, nt * 16, kt * 32, lane), acc);
    store_d_f16(SF, 128, nt * 16, acc, lane);
  }
  // explainer per label
  v16h af[4];
#pragma unroll
  for (int kt = 0; kt < 4; ++kt) af[kt] = load_a_f16(SF, 128, kt * 32, lane);
  float prob = 0.f;
#pragma unroll
  for (int l = 0; l < NLBL; ++l){
    const _Float16* w1 = We1t + (size_t)l * 128 * 128;
    const _Float16* w2 = We2t + (size_t)l * 64 * 128;
    // h1 = elu(feats @ We1[l] + be1[l]) -> SB cols 0..127
#pragma unroll
    for (int nt = 0; nt < 8; ++nt){
      v8f acc = bias_acc(be1 + l * 128, nt * 16, lane);
#pragma unroll
      for (int kt = 0; kt < 4; ++kt)
        acc = WMMA_F16(af[kt], load_b_f16(w1, 128, nt * 16, kt * 32, lane), acc);
      store_d_f16(SB, 256, nt * 16, elu_v(acc), lane);
    }
    // h2 = elu(h1 @ We2[l] + be2[l]) -> SB cols 128..191
    v16h ah[4];
#pragma unroll
    for (int kt = 0; kt < 4; ++kt) ah[kt] = load_a_f16(SB, 256, kt * 32, lane);
#pragma unroll
    for (int nt = 0; nt < 4; ++nt){
      v8f acc = bias_acc(be2 + l * 64, nt * 16, lane);
#pragma unroll
      for (int kt = 0; kt < 4; ++kt)
        acc = WMMA_F16(ah[kt], load_b_f16(w2, 128, nt * 16, kt * 32, lane), acc);
      store_d_f16(SB, 256, 128 + nt * 16, elu_v(acc), lane);
    }
    // score = h2 . We3[l] + be3[l] (split dot across lane halves, combine with shfl)
    const _Float16* hp = SB + r * 256 + 128 + hsel * 32;
    const float* w3 = We3 + l * 64 + hsel * 32;
    float s = 0.f;
#pragma unroll
    for (int i = 0; i < 32; ++i) s += (float)hp[i] * w3[i];
    s += __shfl_xor(s, 16);
    s += be3[l];
    if (label == l) prob = s;
  }
  if (hsel == 0 && (t0 + r) < Ea) probs[t0 + r] = prob;
}

// ---------------- segment max / argmin / finalize ----------------
__global__ void k_segmax(const float* __restrict__ probs, const int* __restrict__ aidx,
                         const int* __restrict__ src, const int* __restrict__ batch,
                         unsigned* __restrict__ segbits, int Ea){
  int t = blockIdx.x * blockDim.x + threadIdx.x;
  if (t >= Ea) return;
  int b = batch[src[aidx[t]]];
  atomicMax(&segbits[b], enc_f32(probs[t]));
}
__global__ void k_action(const float* __restrict__ probs, const int* __restrict__ aidx,
                         const int* __restrict__ src, const int* __restrict__ batch,
                         const unsigned* __restrict__ segbits, int* __restrict__ action, int Ea){
  int t = blockIdx.x * blockDim.x + threadIdx.x;
  if (t >= Ea) return;
  int b = batch[src[aidx[t]]];
  if (enc_f32(probs[t]) == segbits[b]) atomicMin(&action[b], t);
}
__global__ void k_final(const unsigned* __restrict__ segbits, const int* __restrict__ action,
                        float* __restrict__ out_tail){
  int i = threadIdx.x;
  if (i < NB)           out_tail[i] = dec_f32(segbits[i]);
  else if (i < 2 * NB)  out_tail[i] = (float)action[i - NB];
  else if (i < 3 * NB)  out_tail[i] = (float)(i - 2 * NB);
}

extern "C" void kernel_launch(void* const* d_in, const int* in_sizes, int n_in,
                              void* d_out, int out_size, void* d_ws, size_t ws_size,
                              hipStream_t stream){
  const float* x   = (const float*)d_in[0];
  const int* ei    = (const int*)d_in[1];
  const int* batch = (const int*)d_in[2];
  const int* y     = (const int*)d_in[3];
  const unsigned char* state = (const unsigned char*)d_in[4];
  const float* Wm1 = (const float*)d_in[5];
  const float* Wm2 = (const float*)d_in[6];
  const float* Wg  = (const float*)d_in[7];
  const float* W1  = (const float*)d_in[8];
  const float* b1  = (const float*)d_in[9];
  const float* W2  = (const float*)d_in[10];
  const float* b2  = (const float*)d_in[11];
  const float* W3  = (const float*)d_in[12];
  const float* b3  = (const float*)d_in[13];
  const float* We1 = (const float*)d_in[14];
  const float* be1 = (const float*)d_in[15];
  const float* We2 = (const float*)d_in[16];
  const float* be2 = (const float*)d_in[17];
  const float* We3 = (const float*)d_in[18];
  const float* be3 = (const float*)d_in[19];

  int Nn = in_sizes[0] / DIM;       // nodes
  int E  = in_sizes[1] / 2;         // edges
  const int* srcp = ei;
  const int* dstp = ei + E;
  int Ea = out_size - 3 * NB;       // available edges

  char* w = (char*)d_ws;
  auto alloc = [&](size_t bytes)->char*{ char* p = w; w += (bytes + 255) & ~(size_t)255; return p; };
  float* h_buf    = (float*)alloc((size_t)Nn * HID * 4);
  char* zero0 = w;
  float* agg_sf   = (float*)alloc((size_t)Nn * HID * 4);
  float* agg_ava  = (float*)alloc((size_t)Nn * HID * 4);
  float* nmask    = (float*)alloc((size_t)Nn * 4);
  float* sum_ones = (float*)alloc(NB * RDIM * 4);
  float* sum_sf   = (float*)alloc(NB * RDIM * 4);
  float* cnt_ones = (float*)alloc(NB * 4);
  float* cnt_sf   = (float*)alloc(NB * 4);
  unsigned* segbits = (unsigned*)alloc(NB * 4);
  char* zero1 = w;
  int* action   = (int*)alloc(NB * 4);
  float* diffb  = (float*)alloc(NB * HID * 4);
  int* blockcnt = (int*)alloc(4096);
  int* blockoff = (int*)alloc(4096);
  int* aidx     = (int*)alloc((size_t)E * 4);
  float* nr_ava = (float*)alloc((size_t)Nn * RDIM * 4);
  _Float16* Wm1t = (_Float16*)alloc((size_t)HID * DIM * 2);
  _Float16* W1t  = (_Float16*)alloc((size_t)256 * 64 * 2);
  _Float16* W2t  = (_Float16*)alloc((size_t)128 * 256 * 2);
  _Float16* W3t  = (_Float16*)alloc((size_t)64 * 128 * 2);
  _Float16* We1t = (_Float16*)alloc((size_t)NLBL * 128 * 128 * 2);
  _Float16* We2t = (_Float16*)alloc((size_t)NLBL * 64 * 128 * 2);
  (void)ws_size; (void)n_in;

  hipMemsetAsync(zero0, 0, (size_t)(zero1 - zero0), stream);
  hipMemsetAsync(action, 0x7F, NB * 4, stream);   // "infinity" for atomicMin

  auto tr = [&](const float* Wp, _Float16* Wt, int K, int Nc){
    int tot = K * Nc;
    k_transpose_w<<<(tot + 255) / 256, 256, 0, stream>>>(Wp, Wt, K, Nc);
  };
  tr(Wm1, Wm1t, DIM, HID);
  tr(W1, W1t, 64, 256);
  tr(W2, W2t, 256, 128);
  tr(W3, W3t, 128, 64);
  for (int l = 0; l < NLBL; ++l){
    tr(We1 + (size_t)l * 128 * 128, We1t + (size_t)l * 128 * 128, 128, 128);
    tr(We2 + (size_t)l * 128 * 64,  We2t + (size_t)l * 64 * 128, 128, 64);
  }

  int htiles = (Nn + 15) / 16;
  k_h<<<(htiles + 3) / 4, 128, 0, stream>>>(x, Wm1t, h_buf, Nn);

  long stot = (long)E * 16;
  k_scatter<<<(int)((stot + 255) / 256), 256, 0, stream>>>(srcp, dstp, state, h_buf,
                                                           agg_sf, agg_ava, nmask, E);
  k_node<<<(Nn + 7) / 8, 256, 0, stream>>>(h_buf, agg_sf, agg_ava, Wm2, nmask, batch,
                                           nr_ava, sum_ones, sum_sf, cnt_ones, cnt_sf, Nn);
  k_diff<<<NB, HID, 0, stream>>>(sum_ones, sum_sf, cnt_ones, cnt_sf, Wg, diffb);

  int nblk = (E + SCAN_TPB * SCAN_EPT - 1) / (SCAN_TPB * SCAN_EPT);
  k_count<<<nblk, SCAN_TPB, 0, stream>>>(state, blockcnt, E);
  k_scanblocks<<<1, 32, 0, stream>>>(blockcnt, blockoff, nblk);
  k_compact<<<nblk, SCAN_TPB, 0, stream>>>(state, blockoff, aidx, E);

  int ntiles = (Ea + 15) / 16;
  k_edge<<<(ntiles + K6_WAVES - 1) / K6_WAVES, 32 * K6_WAVES, 0, stream>>>(
      aidx, srcp, dstp, batch, y, nr_ava, diffb,
      W1t, b1, W2t, b2, W3t, b3, We1t, be1, We2t, be2, We3, be3,
      (float*)d_out, Ea);

  k_segmax<<<(Ea + 255) / 256, 256, 0, stream>>>((const float*)d_out, aidx, srcp, batch, segbits, Ea);
  k_action<<<(Ea + 255) / 256, 256, 0, stream>>>((const float*)d_out, aidx, srcp, batch, segbits, action, Ea);
  k_final<<<1, 3 * NB, 0, stream>>>(segbits, action, (float*)d_out + Ea);
}
